// LSTMModel_41893111005402
// MI455X (gfx1250) — matrix operand order repaired and checked by reading, not compiled, not run on hardware
//
#include <hip/hip_runtime.h>
#include <hip/hip_bf16.h>
#include <math.h>

// ---------------------------------------------------------------------------
// Model dims (fixed by the reference)
// ---------------------------------------------------------------------------
#define BATCH   1024
#define HID     256
#define VOCAB   1000
#define VPAD    1024      // vocab padded to multiple of 16 (64 tiles)
#define INDIM   14
#define TSTEPS  50
#define G4H     1024      // 4*HID

#define AS1 __attribute__((address_space(1)))
#define AS3 __attribute__((address_space(3)))

typedef __attribute__((ext_vector_type(16))) _Float16 v16h;
typedef __attribute__((ext_vector_type(8)))  _Float16 v8h;
typedef __attribute__((ext_vector_type(8)))  float    v8f;
typedef __attribute__((ext_vector_type(4)))  int      v4i;

// ---------------------------------------------------------------------------
// WMMA fragment loaders (wave32, gfx1250; contiguous k order, same in A and B)
// ---------------------------------------------------------------------------
// A-matrix 16x32 f16: lane L holds row m=L&15; lanes 0-15 carry K=0..15,
// lanes 16-31 carry K=16..31 (contiguous).
__device__ __forceinline__ v16h load_a_frag(const _Float16* __restrict__ base,
                                            int ld, int lane) {
    const int m  = lane & 15;
    const int kh = lane >> 4;          // 0 or 1
    const _Float16* p = base + m * ld + kh * 16;
    v8h lo = *(const v8h*)(p);         // K = kh*16 .. kh*16+7
    v8h hi = *(const v8h*)(p + 8);     // K = kh*16+8 .. kh*16+15
    return __builtin_shufflevector(lo, hi,
        0,1,2,3,4,5,6,7, 8,9,10,11,12,13,14,15);
}

// B-matrix 32x16 f16 read from the LDS-staged weight strip Bs[64][HID]
// (strip is row-major [n][k]): lane L holds column n=L&15; lanes 0-15 carry
// K=0..15, lanes 16-31 carry K=16..31 (contiguous).
__device__ __forceinline__ v16h lds_b_frag(const _Float16* Bs, int rowLocal,
                                           int k0, int lane) {
    const int n  = lane & 15;
    const int kh = lane >> 4;
    return *(const v16h*)(Bs + (rowLocal + n) * HID + k0 + kh * 16);
}

// Stage a 64-row x 256-K f16 weight strip (32 KB) into LDS, using the CDNA5
// async global->LDS DMA path when the toolchain exposes it.
__device__ __forceinline__ void stage_strip(const _Float16* __restrict__ gsrc,
                                            _Float16* Bs, int tid) {
#if __has_builtin(__builtin_amdgcn_global_load_async_to_lds_b128)
    AS1 char* g = (AS1 char*)gsrc;
    AS3 char* l = (AS3 char*)Bs;
    #pragma unroll
    for (int i = 0; i < 8; ++i) {
        const int byteOff = (tid + i * 256) * 16;      // 256 thr * 8 * 16B = 32KB
        __builtin_amdgcn_global_load_async_to_lds_b128(
            (AS1 v4i*)(g + byteOff), (AS3 v4i*)(l + byteOff), 0, 0);
    }
#else
    const v8h* gs = (const v8h*)gsrc;
    v8h*       ls = (v8h*)Bs;
    #pragma unroll
    for (int i = 0; i < 8; ++i)
        ls[tid + i * 256] = gs[tid + i * 256];
#endif
}

__device__ __forceinline__ void wait_stage_done() {
#if __has_builtin(__builtin_amdgcn_global_load_async_to_lds_b128)
#if __has_builtin(__builtin_amdgcn_s_wait_asynccnt)
    __builtin_amdgcn_s_wait_asynccnt(0);
#else
    asm volatile("s_wait_asynccnt 0x0" ::: "memory");
#endif
#endif
    __syncthreads();
}

__device__ __forceinline__ float sigf(float x) {
    return 1.0f / (1.0f + __expf(-x));
}

// ---------------------------------------------------------------------------
// Prep kernels
// ---------------------------------------------------------------------------
__global__ void convert_weights(const float* __restrict__ Whh,
                                const float* __restrict__ Wout,
                                _Float16* __restrict__ whh16,
                                _Float16* __restrict__ wout16) {
    const int idx = blockIdx.x * blockDim.x + threadIdx.x;   // < 1024*256
    whh16[idx] = (_Float16)Whh[idx];
    const int row = idx >> 8;                                 // /HID
    wout16[idx] = (row < VOCAB) ? (_Float16)Wout[idx] : (_Float16)0.0f;
}

__global__ void init_state(_Float16* __restrict__ h16, float* __restrict__ c) {
    const int idx = blockIdx.x * blockDim.x + threadIdx.x;    // < B*H
    h16[idx] = (_Float16)0.0f;
    c[idx]   = 0.0f;
}

// x = inputs @ W_in^T + b_in   [B,H], tiny 14-dot, f32
__global__ void compute_x(const float* __restrict__ inputs,
                          const float* __restrict__ W_in,
                          const float* __restrict__ b_in,
                          float* __restrict__ x) {
    const int idx = blockIdx.x * blockDim.x + threadIdx.x;    // < B*H
    const int b = idx >> 8, j = idx & 255;
    float acc = b_in[j];
    #pragma unroll
    for (int d = 0; d < INDIM; ++d)
        acc = fmaf(inputs[b * INDIM + d], W_in[j * INDIM + d], acc);
    x[idx] = acc;
}

// xg = x @ W_ih^T + (b_ih + b_hh)   [B,4H], f32 (one-time)
__global__ void compute_xg(const float* __restrict__ x,
                           const float* __restrict__ W_ih,
                           const float* __restrict__ b_ih,
                           const float* __restrict__ b_hh,
                           float* __restrict__ xg) {
    const int idx = blockIdx.x * blockDim.x + threadIdx.x;    // < B*4H
    const int b = idx >> 10, n = idx & 1023;
    const float* xr = x + b * HID;
    const float* wr = W_ih + n * HID;
    float acc = b_ih[n] + b_hh[n];
    #pragma unroll 8
    for (int j = 0; j < HID; ++j)
        acc = fmaf(xr[j], wr[j], acc);
    xg[idx] = acc;
}

// ---------------------------------------------------------------------------
// Per-step WMMA GEMMs. Grid = 8 M-groups x 16 N-strips = 128 blocks of 8
// waves. All 8 waves of a block share one 64-col N-strip whose weight panel
// (64x256 f16, 32 KB) is async-DMA'd into LDS once; each wave owns one
// 16-row M tile and computes a 16x64 strip (4 accumulators). Both the A
// fragment (global) and the 4 B fragments (LDS) are software-pipelined one
// k-step ahead so ds/vmem latency overlaps the WMMA issue.
// ---------------------------------------------------------------------------
__global__ void gemm_gates(const _Float16* __restrict__ h16,
                           const _Float16* __restrict__ whh16,
                           const float*    __restrict__ xg,
                           float*          __restrict__ g) {
    __shared__ __align__(32) _Float16 Bs[64 * HID];

    const int tid  = threadIdx.x;
    const int lane = tid & 31;
    const int wave = tid >> 5;                 // 0..7
    const int col0 = (blockIdx.x & 15) * 64;   // N strip
    const int row0 = ((blockIdx.x >> 4) * 8 + wave) * 16;   // M tile

    stage_strip(whh16 + col0 * HID, Bs, tid);  // async -> LDS (overlaps below)

    const int n  = lane & 15;
    const int mb = (lane >> 4) * 8;            // C/D: VGPR v = row mb+v, col n

    v8f acc[4];
    #pragma unroll
    for (int j = 0; j < 4; ++j) {
        const float* cp = xg + (row0 + mb) * G4H + col0 + j * 16 + n;
        #pragma unroll
        for (int v = 0; v < 8; ++v) acc[j][v] = cp[v * G4H];
    }

    v16h a = load_a_frag(h16 + row0 * HID, HID, lane);  // k0 = 0 prefetch
    wait_stage_done();

    v16h bf[4];
    #pragma unroll
    for (int j = 0; j < 4; ++j) bf[j] = lds_b_frag(Bs, j * 16, 0, lane);

    #pragma unroll
    for (int k0 = 0; k0 < HID; k0 += 32) {
        v16h a_next = a;
        v16h bf_next[4];
        if (k0 + 32 < HID) {
            a_next = load_a_frag(h16 + row0 * HID + k0 + 32, HID, lane);
            #pragma unroll
            for (int j = 0; j < 4; ++j)
                bf_next[j] = lds_b_frag(Bs, j * 16, k0 + 32, lane);
        } else {
            #pragma unroll
            for (int j = 0; j < 4; ++j) bf_next[j] = bf[j];
        }
        #pragma unroll
        for (int j = 0; j < 4; ++j)
            acc[j] = __builtin_amdgcn_wmma_f32_16x16x32_f16(
                false, a, false, bf[j], (short)0, acc[j], false, false);
        a = a_next;
        #pragma unroll
        for (int j = 0; j < 4; ++j) bf[j] = bf_next[j];
    }

    #pragma unroll
    for (int j = 0; j < 4; ++j) {
        float* cp = g + (row0 + mb) * G4H + col0 + j * 16 + n;
        #pragma unroll
        for (int v = 0; v < 8; ++v) cp[v * G4H] = acc[j][v];
    }
}

__global__ void gemm_logits(const _Float16* __restrict__ h16,
                            const _Float16* __restrict__ wout16,
                            const float*    __restrict__ b_out,
                            float*          __restrict__ s) {
    __shared__ __align__(32) _Float16 Bs[64 * HID];

    const int tid  = threadIdx.x;
    const int lane = tid & 31;
    const int wave = tid >> 5;
    const int col0 = (blockIdx.x & 15) * 64;
    const int row0 = ((blockIdx.x >> 4) * 8 + wave) * 16;

    stage_strip(wout16 + col0 * HID, Bs, tid);

    const int n  = lane & 15;
    const int mb = (lane >> 4) * 8;

    v8f acc[4];
    #pragma unroll
    for (int j = 0; j < 4; ++j) {
        const int col = col0 + j * 16 + n;
        const float bias = (col < VOCAB) ? b_out[col] : 0.0f;
        #pragma unroll
        for (int v = 0; v < 8; ++v) acc[j][v] = bias;
    }

    v16h a = load_a_frag(h16 + row0 * HID, HID, lane);
    wait_stage_done();

    v16h bf[4];
    #pragma unroll
    for (int j = 0; j < 4; ++j) bf[j] = lds_b_frag(Bs, j * 16, 0, lane);

    #pragma unroll
    for (int k0 = 0; k0 < HID; k0 += 32) {
        v16h a_next = a;
        v16h bf_next[4];
        if (k0 + 32 < HID) {
            a_next = load_a_frag(h16 + row0 * HID + k0 + 32, HID, lane);
            #pragma unroll
            for (int j = 0; j < 4; ++j)
                bf_next[j] = lds_b_frag(Bs, j * 16, k0 + 32, lane);
        } else {
            #pragma unroll
            for (int j = 0; j < 4; ++j) bf_next[j] = bf[j];
        }
        #pragma unroll
        for (int j = 0; j < 4; ++j)
            acc[j] = __builtin_amdgcn_wmma_f32_16x16x32_f16(
                false, a, false, bf[j], (short)0, acc[j], false, false);
        a = a_next;
        #pragma unroll
        for (int j = 0; j < 4; ++j) bf[j] = bf_next[j];
    }

    #pragma unroll
    for (int j = 0; j < 4; ++j) {
        float* cp = s + (row0 + mb) * VPAD + col0 + j * 16 + n;
        #pragma unroll
        for (int v = 0; v < 8; ++v) cp[v * VPAD] = acc[j][v];
    }
}

// ---------------------------------------------------------------------------
// LSTM elementwise update: c = f*c + i*g ; h = o*tanh(c)  (PyTorch i,f,g,o)
// ---------------------------------------------------------------------------
__global__ void lstm_update(const float* __restrict__ g,
                            float* __restrict__ c,
                            _Float16* __restrict__ h16) {
    const int idx = blockIdx.x * blockDim.x + threadIdx.x;    // < B*H
    const int b = idx >> 8, j = idx & 255;
    const float* gr = g + b * G4H;
    const float ig = sigf(gr[j]);
    const float fg = sigf(gr[j + HID]);
    const float gg = tanhf(gr[j + 2 * HID]);
    const float og = sigf(gr[j + 3 * HID]);
    const float cv = fg * c[idx] + ig * gg;
    c[idx]   = cv;
    h16[idx] = (_Float16)(og * tanhf(cv));
}

// ---------------------------------------------------------------------------
// Row softmax over first VOCAB cols of s, write out[:, t, :]
// ---------------------------------------------------------------------------
__global__ void softmax_rows(const float* __restrict__ s,
                             float* __restrict__ out, int t) {
    __shared__ float red[256];
    const int b   = blockIdx.x;
    const int tid = threadIdx.x;
    const float* row = s + b * VPAD;

    float m = -INFINITY;
    for (int v = tid; v < VOCAB; v += 256) m = fmaxf(m, row[v]);
    red[tid] = m;
    __syncthreads();
    for (int off = 128; off > 0; off >>= 1) {
        if (tid < off) red[tid] = fmaxf(red[tid], red[tid + off]);
        __syncthreads();
    }
    const float mx = red[0];
    __syncthreads();

    float sum = 0.0f;
    for (int v = tid; v < VOCAB; v += 256) sum += __expf(row[v] - mx);
    red[tid] = sum;
    __syncthreads();
    for (int off = 128; off > 0; off >>= 1) {
        if (tid < off) red[tid] += red[tid + off];
        __syncthreads();
    }
    const float inv = 1.0f / red[0];

    float* op = out + ((size_t)b * TSTEPS + t) * VOCAB;
    for (int v = tid; v < VOCAB; v += 256) op[v] = __expf(row[v] - mx) * inv;
}

// ---------------------------------------------------------------------------
// Launcher
// ---------------------------------------------------------------------------
extern "C" void kernel_launch(void* const* d_in, const int* in_sizes, int n_in,
                              void* d_out, int out_size, void* d_ws, size_t ws_size,
                              hipStream_t stream) {
    (void)in_sizes; (void)n_in; (void)out_size; (void)ws_size;

    const float* inputs = (const float*)d_in[0];
    const float* W_in   = (const float*)d_in[1];
    const float* b_in   = (const float*)d_in[2];
    const float* W_ih   = (const float*)d_in[3];
    const float* b_ih   = (const float*)d_in[4];
    const float* W_hh   = (const float*)d_in[5];
    const float* b_hh   = (const float*)d_in[6];
    const float* W_out  = (const float*)d_in[7];
    const float* b_out  = (const float*)d_in[8];
    float* out = (float*)d_out;

    // Workspace layout (bytes), total ~11.5 MB
    char* ws = (char*)d_ws;
    _Float16* whh16  = (_Float16*)(ws);                       // 1024*256*2 = 512K
    _Float16* wout16 = (_Float16*)(ws + (512u << 10));        // 512K
    _Float16* h16    = (_Float16*)(ws + (1024u << 10));       // 512K
    float*    c      = (float*)   (ws + (1536u << 10));       // 1M
    float*    x      = (float*)   (ws + (2560u << 10));       // 1M
    float*    xg     = (float*)   (ws + (3584u << 10));       // 4M
    float*    scr    = (float*)   (ws + (7680u << 10));       // 4M (gates & logits)

    // One-time prep
    convert_weights<<<BATCH, 256, 0, stream>>>(W_hh, W_out, whh16, wout16);
    init_state<<<(BATCH * HID) / 256, 256, 0, stream>>>(h16, c);
    compute_x<<<(BATCH * HID) / 256, 256, 0, stream>>>(inputs, W_in, b_in, x);
    compute_xg<<<(BATCH * G4H) / 256, 256, 0, stream>>>(x, W_ih, b_ih, b_hh, xg);

    // 50 recurrent steps: gates -> update -> logits -> softmax
    for (int t = 0; t < TSTEPS; ++t) {
        gemm_gates <<<128, 256, 0, stream>>>(h16, whh16, xg, scr);
        lstm_update<<<(BATCH * HID) / 256, 256, 0, stream>>>(scr, c, h16);
        gemm_logits<<<128, 256, 0, stream>>>(h16, wout16, b_out, scr);
        softmax_rows<<<BATCH, 256, 0, stream>>>(scr, out, t);
    }
}
